// TransformerEncoderLayer_3461743640605
// MI455X (gfx1250) — compile-verified, matrix operand order checked by reading, not run on hardware
//
#include <hip/hip_runtime.h>
#include <hip/hip_bf16.h>

// ---------------------------------------------------------------------------
// Problem constants (match reference)
// ---------------------------------------------------------------------------
constexpr int Bc  = 4;
constexpr int Lc  = 2048;
constexpr int Dc  = 1024;
constexpr int Hc  = 16;
constexpr int DFFc = 4096;
constexpr int HDc = Dc / Hc;           // 64
constexpr int Mrows = Bc * Lc;         // 8192
constexpr int MAXDc = 512;             // rel bias table has 2*512-1 = 1023 rows

// ---------------------------------------------------------------------------
// Feature probes (compile-only environment: gate every uncertain builtin)
// ---------------------------------------------------------------------------
#if defined(__has_builtin)
# if __has_builtin(__builtin_amdgcn_tensor_load_to_lds)
#  define HAVE_TDM 1
# endif
# if __has_builtin(__builtin_amdgcn_global_load_async_to_lds_b128)
#  define HAVE_ASYNC 1
# endif
#endif

// ---------------------------------------------------------------------------
// WMMA types (CDNA5 / gfx1250, wave32)
// ---------------------------------------------------------------------------
typedef __attribute__((ext_vector_type(16))) _Float16 v16h;
typedef __attribute__((ext_vector_type(8)))  _Float16 v8h;
typedef __attribute__((ext_vector_type(8)))  float    v8f;
typedef __attribute__((ext_vector_type(4)))  unsigned int v4u;
typedef __attribute__((ext_vector_type(8)))  int v8i;
typedef __attribute__((ext_vector_type(4)))  int v4i;

union V16 { v16h v; v8h h[2]; };

__device__ __forceinline__ v8f wmma_f32_16x16x32(v16h a, v16h b, v8f c) {
    return __builtin_amdgcn_wmma_f32_16x16x32_f16(
        /*neg_a=*/false, a, /*neg_b=*/false, b,
        /*c_mod=*/(short)0, c, /*reuse_a=*/false, /*reuse_b=*/false);
}

// ---------------------------------------------------------------------------
// gfx1250 data-mover helpers (guarded)
// ---------------------------------------------------------------------------
#if defined(HAVE_ASYNC)
// Parameter types per hipcc diagnostic: (int4 AS1*, int4 AS3*, imm, imm)
typedef __attribute__((address_space(1))) v4i gv4i;
typedef __attribute__((address_space(3))) v4i lv4i;
__device__ __forceinline__ void async_copy16(const void* g, void* l) {
    // GLOBAL_LOAD_ASYNC_TO_LDS_B128: 16B per lane, tracked by ASYNCcnt
    __builtin_amdgcn_global_load_async_to_lds_b128(
        (gv4i*)(uintptr_t)g,
        (lv4i*)(unsigned int)(uintptr_t)l, 0, 0);
}
__device__ __forceinline__ void wait_async0() {
# if __has_builtin(__builtin_amdgcn_s_wait_asynccnt)
    __builtin_amdgcn_s_wait_asynccnt(0);
# else
    asm volatile("s_wait_asynccnt 0x0" ::: "memory");
# endif
}
#endif

#if defined(HAVE_TDM)
__device__ __forceinline__ void wait_tensor0() {
# if __has_builtin(__builtin_amdgcn_s_wait_tensorcnt)
    __builtin_amdgcn_s_wait_tensorcnt(0);
# else
    asm volatile("s_wait_tensorcnt 0x0" ::: "memory");
# endif
}

// 2D TDM tile load: tileH rows x tileW f16, row stride = rowStrideElems.
// pad fields reproduce the LDS row padding used by the fragment readers.
__device__ __forceinline__ void tdm_load_2d_f16(const _Float16* gsrc, void* ldsDst,
                                                unsigned int tileW, unsigned int tileH,
                                                unsigned int rowStrideElems,
                                                unsigned int padIntervalCode,
                                                unsigned int padAmountCode,
                                                bool padEn) {
    unsigned long long ga = (unsigned long long)(uintptr_t)gsrc;
    unsigned int lds = (unsigned int)(uintptr_t)ldsDst;   // low 32 bits = LDS offset
    v4u g0 = { 0x1u /*count=1*/, lds, (unsigned int)ga,
               (unsigned int)((ga >> 32) & 0x01FFFFFFu) | (2u << 30) /*type=2*/ };
    unsigned int w0 = (1u << 16) /*data_size=2B*/
                    | (padEn ? (1u << 20) : 0u)
                    | (padIntervalCode << 22) | (padAmountCode << 25);
    unsigned int td0 = rowStrideElems;        // tensor_dim0 (elems)
    unsigned int td1 = 0x7FFFFFFFu;           // tensor_dim1: effectively no clip
    unsigned long long s0 = rowStrideElems;   // tensor_dim0_stride
    v8i g1 = { (int)w0,
               (int)((td0 & 0xFFFFu) << 16),
               (int)((td0 >> 16) | ((td1 & 0xFFFFu) << 16)),
               (int)((td1 >> 16) | (tileW << 16)),
               (int)tileH,
               (int)(unsigned int)s0,
               (int)(unsigned int)(s0 >> 32),
               0 };
    v4i z = { 0, 0, 0, 0 };
# if defined(__clang_major__) && (__clang_major__ >= 23)
    v8i z8 = { 0, 0, 0, 0, 0, 0, 0, 0 };
    __builtin_amdgcn_tensor_load_to_lds(g0, g1, z, z, z8, 0);
# else
    __builtin_amdgcn_tensor_load_to_lds(g0, g1, z, z, 0);
# endif
}
#endif

__device__ __forceinline__ void wait_lds_wave_local() {
#if defined(__has_builtin) && __has_builtin(__builtin_amdgcn_s_wait_dscnt)
    __builtin_amdgcn_s_wait_dscnt(0);
#else
    asm volatile("s_wait_dscnt 0x0" ::: "memory");
#endif
}

// ---------------------------------------------------------------------------
// fp32 -> fp16 conversion (grid-stride)
// ---------------------------------------------------------------------------
__global__ void cvt_f32_to_f16_kernel(const float* __restrict__ src,
                                      _Float16* __restrict__ dst, int n) {
    int i = blockIdx.x * blockDim.x + threadIdx.x;
    int stride = gridDim.x * blockDim.x;
    for (; i < n; i += stride) dst[i] = (_Float16)src[i];
}

// ---------------------------------------------------------------------------
// Tiled WMMA GEMM: C[M,N] = A[M,K] * B[K,N]   (A,B fp16 row-major, acc fp32)
// 256 threads = 8 waves; tile 128x128x32; LDS double-buffered so the next
// tile's staging (TDM / async / manual) overlaps the 8 WMMAs of this tile.
// EPI: 0 = store f16 raw; 1 = f32: acc+bias+residual; 2 = f16: gelu(acc+bias)
// ---------------------------------------------------------------------------
template <int EPI>
__global__ __launch_bounds__(256)
void gemm_f16_wmma_kernel(const _Float16* __restrict__ A,
                          const _Float16* __restrict__ Bm,
                          float* __restrict__ outF,
                          _Float16* __restrict__ outH,
                          const float* __restrict__ bias,
                          const float* __restrict__ residual,
                          int M, int N, int K) {
    constexpr int BM = 128, BN = 128, BK = 32;
    constexpr int LDS_S = BK + 16;   // 48 elems = 96B rows (16B aligned)
    __shared__ _Float16 ldsA[2][BM * LDS_S];
    __shared__ _Float16 ldsB[2][BN * LDS_S];   // transposed: [n][k]

    const int t    = threadIdx.x;
    const int lane = t & 31;
    const int w    = t >> 5;
    const int wm   = w >> 2;       // 0..1
    const int wn   = w & 3;        // 0..3
    const int bm   = blockIdx.y * BM;
    const int bn   = blockIdx.x * BN;
    const int lr   = lane & 15;
    const int kb   = (lane >> 4) * 8;

    v8f acc[4][2] = {};

    auto stageA = [&](int buf, int k0) {
#if defined(HAVE_TDM)
        // One wave-op DMA: 128 rows x 32 f16, hardware-padded 64B data + 32B pad
        // per row -> 96B LDS row stride (= LDS_S elems). TENSORcnt tracked.
        if (t == 0)
            tdm_load_2d_f16(&A[(size_t)bm * K + k0], &ldsA[buf][0],
                            /*tileW=*/BK, /*tileH=*/BM, /*rowStride=*/(unsigned)K,
                            /*intervalCode 16DW=*/3u, /*amountCode 8DW=*/7u, true);
#elif defined(HAVE_ASYNC)
        int r = (t >> 2), c = (t & 3) * 8;
        async_copy16(&A[(size_t)(bm + r) * K + k0 + c], &ldsA[buf][r * LDS_S + c]);
        async_copy16(&A[(size_t)(bm + r + 64) * K + k0 + c],
                     &ldsA[buf][(r + 64) * LDS_S + c]);
#else
        int r = (t >> 2), c = (t & 3) * 8;
        *(v8h*)&ldsA[buf][r * LDS_S + c] =
            *(const v8h*)&A[(size_t)(bm + r) * K + k0 + c];
        *(v8h*)&ldsA[buf][(r + 64) * LDS_S + c] =
            *(const v8h*)&A[(size_t)(bm + r + 64) * K + k0 + c];
#endif
    };

    auto stageB = [&](int buf, int k0) {
        int kr = t >> 3, c = (t & 7) * 16;
        const _Float16* src = &Bm[(size_t)(k0 + kr) * N + bn + c];
        v8h b0 = *(const v8h*)src;
        v8h b1 = *(const v8h*)(src + 8);
#pragma unroll
        for (int i = 0; i < 8; ++i) {
            ldsB[buf][(c + i) * LDS_S + kr]     = b0[i];
            ldsB[buf][(c + 8 + i) * LDS_S + kr] = b1[i];
        }
    };

    auto waitStage = [&]() {
#if defined(HAVE_TDM)
        wait_tensor0();
#elif defined(HAVE_ASYNC)
        wait_async0();
#endif
    };

    // prologue
    stageB(0, 0);
    stageA(0, 0);
    waitStage();
    __syncthreads();

    for (int k0 = 0; k0 < K; k0 += BK) {
        const int cur = (k0 / BK) & 1, nxt = cur ^ 1;
        const bool more = (k0 + BK) < K;
        if (more) {
            stageB(nxt, k0 + BK);
            stageA(nxt, k0 + BK);
        }
        if (k0 + 2 * BK < K) {
            __builtin_prefetch(&A[(size_t)(bm + (t >> 2)) * K + k0 + 2 * BK], 0, 1);
            __builtin_prefetch(&Bm[(size_t)(k0 + 2 * BK + (t >> 3)) * N + bn + (t & 7) * 16], 0, 1);
        }

        V16 aF[4], bF[2];
#pragma unroll
        for (int mt = 0; mt < 4; ++mt) {
            const _Float16* p = &ldsA[cur][(wm * 64 + mt * 16 + lr) * LDS_S + kb];
            aF[mt].h[0] = *(const v8h*)p;
            aF[mt].h[1] = *(const v8h*)(p + 16);
        }
#pragma unroll
        for (int nt = 0; nt < 2; ++nt) {
            const _Float16* p = &ldsB[cur][(wn * 32 + nt * 16 + lr) * LDS_S + kb];
            bF[nt].h[0] = *(const v8h*)p;
            bF[nt].h[1] = *(const v8h*)(p + 16);
        }
#pragma unroll
        for (int mt = 0; mt < 4; ++mt)
#pragma unroll
            for (int nt = 0; nt < 2; ++nt)
                acc[mt][nt] = wmma_f32_16x16x32(aF[mt].v, bF[nt].v, acc[mt][nt]);

        waitStage();          // next tile's DMA finished under the WMMAs
        __syncthreads();
    }

    // epilogue: C layout: lanes 0-15 -> M=j, lanes 16-31 -> M=j+8; N=lane&15
    const int hi = lane >> 4;
#pragma unroll
    for (int mt = 0; mt < 4; ++mt) {
#pragma unroll
        for (int nt = 0; nt < 2; ++nt) {
#pragma unroll
            for (int j = 0; j < 8; ++j) {
                int row = bm + wm * 64 + mt * 16 + j + hi * 8;
                int col = bn + wn * 32 + nt * 16 + lr;
                size_t idx = (size_t)row * N + col;
                float v = acc[mt][nt][j];
                if (EPI == 0) {
                    outH[idx] = (_Float16)v;
                } else if (EPI == 1) {
                    outF[idx] = v + bias[col] + residual[idx];
                } else {
                    float z = v + bias[col];
                    float g = 0.5f * z * (1.0f + erff(z * 0.70710678118654752f));
                    outH[idx] = (_Float16)g;
                }
            }
        }
    }
}

// ---------------------------------------------------------------------------
// Flash attention with relative-position bias.
// Grid: (L/64, H, B). Block: 128 threads = 4 waves; wave w owns 16 query rows.
// ---------------------------------------------------------------------------
__global__ __launch_bounds__(128)
void attn_flash_wmma_kernel(const _Float16* __restrict__ qkv,
                            const float* __restrict__ bias_table,
                            _Float16* __restrict__ ctx) {
    constexpr int LK = HDc + 8;          // 72 elems = 144B rows (16B aligned)
    __shared__ _Float16 ldsK[64 * LK];   // [key][hd]
    __shared__ _Float16 ldsVt[64 * LK];  // [hd][key]  (transposed V)
    __shared__ _Float16 ldsP[64 * LK];   // [q_local][key]

    const int qb   = blockIdx.x * 64;
    const int h    = blockIdx.y;
    const int b    = blockIdx.z;
    const int t    = threadIdx.x;
    const int lane = t & 31;
    const int w    = t >> 5;
    const int lr   = lane & 15;
    const int hi   = lane >> 4;
    const int kb   = hi * 8;
    const float scale = 0.125f;          // HD^-0.5 = 1/8

    // Q fragments for this wave's 16 rows (A operand, 2 chunks over HD=64)
    V16 aQ[2];
    {
        int q = qb + w * 16 + lr;
        const _Float16* p = &qkv[(size_t)(b * Lc + q) * (3 * Dc) + h * HDc];
#pragma unroll
        for (int kc = 0; kc < 2; ++kc) {
            aQ[kc].h[0] = *(const v8h*)(p + kc * 32 + kb);
            aQ[kc].h[1] = *(const v8h*)(p + kc * 32 + kb + 16);
        }
    }

    v8f accO[4] = {};
    float mstat[8], lstat[8];
#pragma unroll
    for (int j = 0; j < 8; ++j) { mstat[j] = -1e30f; lstat[j] = 0.0f; }

    for (int kt = 0; kt < Lc / 64; ++kt) {
        const int k0 = kt * 64;
        __syncthreads();   // previous iteration's LDS reads complete

        // ---- stage K tile [64key x 64hd] (async if available) and V^T tile
        {
            int kl = t >> 1, c = (t & 1) * 32;
            const _Float16* pk =
                &qkv[(size_t)(b * Lc + k0 + kl) * (3 * Dc) + Dc + h * HDc + c];
            const _Float16* pv =
                &qkv[(size_t)(b * Lc + k0 + kl) * (3 * Dc) + 2 * Dc + h * HDc + c];
#if defined(HAVE_ASYNC)
#pragma unroll
            for (int u = 0; u < 4; ++u)
                async_copy16(pk + u * 8, &ldsK[kl * LK + c + u * 8]);
#else
#pragma unroll
            for (int u = 0; u < 4; ++u)
                *(v8h*)&ldsK[kl * LK + c + u * 8] = *(const v8h*)(pk + u * 8);
#endif
#pragma unroll
            for (int u = 0; u < 4; ++u) {
                v8h vv = *(const v8h*)(pv + u * 8);
#pragma unroll
                for (int i = 0; i < 8; ++i)
                    ldsVt[(c + u * 8 + i) * LK + kl] = vv[i];
            }
        }
#if defined(HAVE_ASYNC)
        wait_async0();
#endif
        __syncthreads();

        // ---- S = Q K^T  (16q x 64key per wave): 4 ntiles x 2 hd-chunks
        v8f accS[4] = {};
#pragma unroll
        for (int nt = 0; nt < 4; ++nt) {
#pragma unroll
            for (int kc = 0; kc < 2; ++kc) {
                V16 bK;
                const _Float16* p = &ldsK[(nt * 16 + lr) * LK + kc * 32 + kb];
                bK.h[0] = *(const v8h*)p;
                bK.h[1] = *(const v8h*)(p + 16);
                accS[nt] = wmma_f32_16x16x32(aQ[kc].v, bK.v, accS[nt]);
            }
        }

        // ---- scale + relative position bias
#pragma unroll
        for (int nt = 0; nt < 4; ++nt) {
#pragma unroll
            for (int j = 0; j < 8; ++j) {
                int q   = qb + w * 16 + j + hi * 8;
                int key = k0 + nt * 16 + lr;
                int rel = key - q + (MAXDc - 1);
                rel = rel < 0 ? 0 : (rel > 2 * MAXDc - 2 ? 2 * MAXDc - 2 : rel);
                accS[nt][j] = accS[nt][j] * scale + bias_table[rel * Hc + h];
            }
        }

        // ---- online softmax (row reductions within 16-lane halves)
        float alpha[8];
#pragma unroll
        for (int j = 0; j < 8; ++j) {
            float m = fmaxf(fmaxf(accS[0][j], accS[1][j]),
                            fmaxf(accS[2][j], accS[3][j]));
#pragma unroll
            for (int x = 1; x < 16; x <<= 1) m = fmaxf(m, __shfl_xor(m, x, 32));
            float mn = fmaxf(mstat[j], m);
            alpha[j] = __expf(mstat[j] - mn);
            mstat[j] = mn;
        }
#pragma unroll
        for (int nt = 0; nt < 4; ++nt) {
#pragma unroll
            for (int j = 0; j < 8; ++j) {
                float pv = __expf(accS[nt][j] - mstat[j]);
                accS[nt][j] = pv;
                ldsP[(w * 16 + j + hi * 8) * LK + nt * 16 + lr] = (_Float16)pv;
            }
        }
#pragma unroll
        for (int j = 0; j < 8; ++j) {
            float rs = accS[0][j] + accS[1][j] + accS[2][j] + accS[3][j];
#pragma unroll
            for (int x = 1; x < 16; x <<= 1) rs += __shfl_xor(rs, x, 32);
            lstat[j] = lstat[j] * alpha[j] + rs;
        }
#pragma unroll
        for (int ot = 0; ot < 4; ++ot)
#pragma unroll
            for (int j = 0; j < 8; ++j) accO[ot][j] *= alpha[j];

        // P tile is wave-private in LDS: only need this wave's DS ops complete
        wait_lds_wave_local();

        // ---- O += P V : A = P (16q x 32key chunks), B = V (32key x 16hd)
#pragma unroll
        for (int kc = 0; kc < 2; ++kc) {
            V16 aP;
            const _Float16* pp = &ldsP[(w * 16 + lr) * LK + kc * 32 + kb];
            aP.h[0] = *(const v8h*)pp;
            aP.h[1] = *(const v8h*)(pp + 16);
#pragma unroll
            for (int ot = 0; ot < 4; ++ot) {
                V16 bV;
                const _Float16* pv = &ldsVt[(ot * 16 + lr) * LK + kc * 32 + kb];
                bV.h[0] = *(const v8h*)pv;
                bV.h[1] = *(const v8h*)(pv + 16);
                accO[ot] = wmma_f32_16x16x32(aP.v, bV.v, accO[ot]);
            }
        }
    }

    // ---- finalize: O /= l, store ctx f16 [B,L,D]
#pragma unroll
    for (int ot = 0; ot < 4; ++ot) {
#pragma unroll
        for (int j = 0; j < 8; ++j) {
            int q = qb + w * 16 + j + hi * 8;
            int c = ot * 16 + lr;
            float o = accO[ot][j] / lstat[j];
            ctx[(size_t)(b * Lc + q) * Dc + h * HDc + c] = (_Float16)o;
        }
    }
}

// ---------------------------------------------------------------------------
// LayerNorm over last dim (D=1024); one row per block (256 threads).
// ---------------------------------------------------------------------------
__global__ __launch_bounds__(256)
void layernorm_kernel(const float* __restrict__ in,
                      const float* __restrict__ gamma,
                      const float* __restrict__ beta,
                      float* __restrict__ outF,
                      _Float16* __restrict__ outH, int Dn) {
    __shared__ float red[2][8];
    const int row = blockIdx.x;
    const float* p = in + (size_t)row * Dn;
    float s = 0.0f, s2 = 0.0f;
    for (int c = threadIdx.x; c < Dn; c += blockDim.x) {
        float v = p[c];
        s += v; s2 += v * v;
    }
#pragma unroll
    for (int m = 1; m < 32; m <<= 1) {
        s  += __shfl_xor(s,  m, 32);
        s2 += __shfl_xor(s2, m, 32);
    }
    const int lane = threadIdx.x & 31, w = threadIdx.x >> 5;
    if (lane == 0) { red[0][w] = s; red[1][w] = s2; }
    __syncthreads();
    if (w == 0) {
        s  = (lane < 8) ? red[0][lane] : 0.0f;
        s2 = (lane < 8) ? red[1][lane] : 0.0f;
#pragma unroll
        for (int m = 1; m < 8; m <<= 1) {
            s  += __shfl_xor(s,  m, 32);
            s2 += __shfl_xor(s2, m, 32);
        }
        if (lane == 0) { red[0][0] = s; red[1][0] = s2; }
    }
    __syncthreads();
    const float mu  = red[0][0] / (float)Dn;
    const float var = red[1][0] / (float)Dn - mu * mu;
    const float rs  = rsqrtf(var + 1e-5f);
    for (int c = threadIdx.x; c < Dn; c += blockDim.x) {
        float v = (p[c] - mu) * rs * gamma[c] + beta[c];
        if (outF) outF[(size_t)row * Dn + c] = v;
        if (outH) outH[(size_t)row * Dn + c] = (_Float16)v;
    }
}

// ---------------------------------------------------------------------------
// Host orchestration
// ---------------------------------------------------------------------------
extern "C" void kernel_launch(void* const* d_in, const int* in_sizes, int n_in,
                              void* d_out, int out_size, void* d_ws, size_t ws_size,
                              hipStream_t stream) {
    const float* x          = (const float*)d_in[0];
    const float* w_qkv      = (const float*)d_in[1];
    const float* w_out      = (const float*)d_in[2];
    const float* b_out      = (const float*)d_in[3];
    const float* bias_table = (const float*)d_in[4];
    const float* gamma1     = (const float*)d_in[5];
    const float* beta1      = (const float*)d_in[6];
    const float* w_ff1      = (const float*)d_in[7];
    const float* b_ff1      = (const float*)d_in[8];
    const float* w_ff2      = (const float*)d_in[9];
    const float* b_ff2      = (const float*)d_in[10];
    const float* gamma2     = (const float*)d_in[11];
    const float* beta2      = (const float*)d_in[12];
    float* out = (float*)d_out;

    char* ws = (char*)d_ws;
    size_t off = 0;
    auto take = [&](size_t bytes) -> void* {
        void* p = ws + off;
        off = (off + bytes + 255) & ~(size_t)255;
        return p;
    };

    _Float16* xh     = (_Float16*)take((size_t)Mrows * Dc * 2);
    _Float16* wqkvh  = (_Float16*)take((size_t)Dc * 3 * Dc * 2);
    _Float16* wouth  = (_Float16*)take((size_t)Dc * Dc * 2);
    _Float16* wff1h  = (_Float16*)take((size_t)Dc * DFFc * 2);
    _Float16* wff2h  = (_Float16*)take((size_t)DFFc * Dc * 2);
    _Float16* qkvh   = (_Float16*)take((size_t)Mrows * 3 * Dc * 2);
    _Float16* ctxh   = (_Float16*)take((size_t)Mrows * Dc * 2);
    float*    y1     = (float*)   take((size_t)Mrows * Dc * 4);  // reused for y2
    float*    xn1f   = (float*)   take((size_t)Mrows * Dc * 4);
    _Float16* xn1h   = (_Float16*)take((size_t)Mrows * Dc * 2);
    _Float16* hh     = (_Float16*)take((size_t)Mrows * DFFc * 2);
    (void)ws_size; (void)n_in; (void)in_sizes; (void)out_size;

    const dim3 cvtGrid(2048), cvtBlk(256);
    cvt_f32_to_f16_kernel<<<cvtGrid, cvtBlk, 0, stream>>>(x,     xh,    Mrows * Dc);
    cvt_f32_to_f16_kernel<<<cvtGrid, cvtBlk, 0, stream>>>(w_qkv, wqkvh, Dc * 3 * Dc);
    cvt_f32_to_f16_kernel<<<cvtGrid, cvtBlk, 0, stream>>>(w_out, wouth, Dc * Dc);
    cvt_f32_to_f16_kernel<<<cvtGrid, cvtBlk, 0, stream>>>(w_ff1, wff1h, Dc * DFFc);
    cvt_f32_to_f16_kernel<<<cvtGrid, cvtBlk, 0, stream>>>(w_ff2, wff2h, DFFc * Dc);

    // QKV projection: [8192,1024] x [1024,3072] -> f16
    gemm_f16_wmma_kernel<0><<<dim3(3 * Dc / 128, Mrows / 128), 256, 0, stream>>>(
        xh, wqkvh, nullptr, qkvh, nullptr, nullptr, Mrows, 3 * Dc, Dc);

    // Attention with relative position bias
    attn_flash_wmma_kernel<<<dim3(Lc / 64, Hc, Bc), 128, 0, stream>>>(
        qkvh, bias_table, ctxh);

    // Output projection + b_out + residual(x) -> y1 (fp32)
    gemm_f16_wmma_kernel<1><<<dim3(Dc / 128, Mrows / 128), 256, 0, stream>>>(
        ctxh, wouth, y1, nullptr, b_out, x, Mrows, Dc, Dc);

    // LN1 -> xn1 (fp32 residual source + fp16 GEMM operand)
    layernorm_kernel<<<Mrows, 256, 0, stream>>>(y1, gamma1, beta1, xn1f, xn1h, Dc);

    // FFN1 + bias + exact GELU -> h (f16)
    gemm_f16_wmma_kernel<2><<<dim3(DFFc / 128, Mrows / 128), 256, 0, stream>>>(
        xn1h, wff1h, nullptr, hh, b_ff1, nullptr, Mrows, DFFc, Dc);

    // FFN2 + b_ff2 + residual(xn1) -> y2 (reuse y1 buffer)
    gemm_f16_wmma_kernel<1><<<dim3(Dc / 128, Mrows / 128), 256, 0, stream>>>(
        hh, wff2h, y1, nullptr, b_ff2, xn1f, Mrows, Dc, DFFc);

    // LN2 -> final output (fp32)
    layernorm_kernel<<<Mrows, 256, 0, stream>>>(y1, gamma2, beta2, out, nullptr, Dc);
}
